// ArcFaceLoss_75677323755947
// MI455X (gfx1250) — matrix operand order repaired and checked by reading, not compiled, not run on hardware
//
#include <hip/hip_runtime.h>

// ---------------- problem constants ----------------
#define BDIM   2048
#define DDIM   128
#define CDIM   100000
#define SCALE_ 64.0f
#define EPS_   1e-7f
// cos(0.5), sin(0.5) for the margin identity cos(acos(t)+M) = t*cosM - sqrt(1-t^2)*sinM
#define COS_M  0.8775825618903728f
#define SIN_M  0.4794255386042030f

// ---------------- tiling ----------------
#define CHUNK  2048
#define NCHUNK ((CDIM + CHUNK - 1) / CHUNK)   // 49

// ---------------- workspace layout (bytes) ----------------
// Wt_hi : [CDIM][DDIM] bf16  (column-normalized W, transposed, hi part)
// Wt_lo : same, residual part
#define OFF_HI     0
#define OFF_LO     (25600000)
#define OFF_INVN   (51200000)
#define OFF_PART   (51600000)                       // float2[BDIM][NCHUNK]
#define OFF_RLOSS  (OFF_PART + BDIM * NCHUNK * 8)   // float[BDIM]

typedef __bf16 bf16_t;
typedef __attribute__((ext_vector_type(16))) __bf16 v16bf;
typedef __attribute__((ext_vector_type(8)))  __bf16 v8bf;
typedef __attribute__((ext_vector_type(8)))  float  v8f;

__device__ __forceinline__ void lse_combine(float& m, float& s, float om, float os) {
    float nm = fmaxf(m, om);
    s = s * __expf(m - nm) + os * __expf(om - nm);
    m = nm;
}

// ============================================================
// Kernel 1: column norms + transposed bf16 hi/lo pre-scaled W
// ============================================================
__global__ __launch_bounds__(256)
void arc_prep(const float* __restrict__ W,
              unsigned short* __restrict__ wtHi,
              unsigned short* __restrict__ wtLo,
              float* __restrict__ invn) {
    int c = blockIdx.x * blockDim.x + threadIdx.x;
    if (c >= CDIM) return;
    float v[DDIM];
    float ss = 0.f;
#pragma unroll
    for (int d = 0; d < DDIM; ++d) {
        v[d] = W[(size_t)d * CDIM + c];          // coalesced across threads
        ss = fmaf(v[d], v[d], ss);
    }
    float inv = 1.0f / fmaxf(sqrtf(ss), 1e-12f);
    invn[c] = inv;
    v8bf hi[DDIM / 8], lo[DDIM / 8];
#pragma unroll
    for (int d = 0; d < DDIM; ++d) {
        float w  = v[d] * inv;
        bf16_t h = (bf16_t)w;
        float  r = w - (float)h;
        hi[d >> 3][d & 7] = h;
        lo[d >> 3][d & 7] = (bf16_t)r;
    }
    v8bf* hp = (v8bf*)(wtHi + (size_t)c * DDIM);
    v8bf* lp = (v8bf*)(wtLo + (size_t)c * DDIM);
#pragma unroll
    for (int i = 0; i < DDIM / 8; ++i) { hp[i] = hi[i]; lp[i] = lo[i]; }
}

// ============================================================
// Kernel 2: fused GEMM (bf16x3 split-precision WMMA) + online
//           log-sum-exp over a 2048-column chunk, 32 rows / WG
// ============================================================
__global__ __launch_bounds__(256)
void arc_gemm_lse(const float* __restrict__ emb,
                  const unsigned short* __restrict__ wtHi,
                  const unsigned short* __restrict__ wtLo,
                  float2* __restrict__ partial) {
    const int lane = threadIdx.x & 31;
    const int wave = threadIdx.x >> 5;
    const int lm   = lane & 15;     // N index within tile / M index for A
    const int lh   = lane >> 4;     // half-wave select
    const int rowBase   = blockIdx.x * 32;
    const int chunkBase = blockIdx.y * CHUNK;
    const int colsInChunk = min((int)CHUNK, CDIM - chunkBase);
    const int nIter = (colsInChunk + 127) >> 7;

    // ---- A fragments (16x32, contiguous k = 16*lh + s, same order as B), hi/lo split, K=128 = 4 ksteps
    v16bf aHi[2][4], aLo[2][4];
#pragma unroll
    for (int t = 0; t < 2; ++t) {
        const float* rowPtr = emb + (size_t)(rowBase + t * 16 + lm) * DDIM;
#pragma unroll
        for (int kk = 0; kk < 4; ++kk) {
#pragma unroll
            for (int r = 0; r < 8; ++r) {
                // halves (2r,2r+1) hold K = base, base+1 (contiguous in emb row)
                int K0 = kk * 32 + 16 * lh + 2 * r;
                float2 f = *(const float2*)(rowPtr + K0);
                bf16_t h0 = (bf16_t)f.x, h1 = (bf16_t)f.y;
                aHi[t][kk][2 * r]     = h0;
                aHi[t][kk][2 * r + 1] = h1;
                aLo[t][kk][2 * r]     = (bf16_t)(f.x - (float)h0);
                aLo[t][kk][2 * r + 1] = (bf16_t)(f.y - (float)h1);
            }
        }
    }

    // ---- per-slot online softmax state (slot = fixed row, streaming columns)
    float m[2][8], s[2][8];
#pragma unroll
    for (int t = 0; t < 2; ++t)
#pragma unroll
        for (int r = 0; r < 8; ++r) { m[t][r] = -1e30f; s[t][r] = 0.f; }

    for (int it = 0; it < nIter; ++it) {
        const int col = chunkBase + it * 128 + wave * 16 + lm;
        const int ccl = min(col, CDIM - 1);
        const unsigned short* hp = wtHi + (size_t)ccl * DDIM;
        const unsigned short* lp = wtLo + (size_t)ccl * DDIM;

        if (it + 1 < nIter) {  // prefetch next B tile (global_prefetch_b8)
            int pc = min(col + 128, CDIM - 1);
            __builtin_prefetch(wtHi + (size_t)pc * DDIM, 0, 1);
            __builtin_prefetch(wtLo + (size_t)pc * DDIM, 0, 1);
        }

        v8f acc0 = {}, acc1 = {};
#pragma unroll
        for (int kk = 0; kk < 4; ++kk) {
            // B fragment (32x16, same contiguous k order as A): lane half lh selects K 0-15 / 16-31,
            // contiguous in the transposed store -> one 32B load each
            const int off = kk * 32 + lh * 16;
            v16bf bHi = *(const v16bf*)(hp + off);
            v16bf bLo = *(const v16bf*)(lp + off);
            // fp32-emulated product: hi*hi + hi*lo + lo*hi
            acc0 = __builtin_amdgcn_wmma_f32_16x16x32_bf16(false, aHi[0][kk], false, bHi, (short)0, acc0, false, false);
            acc1 = __builtin_amdgcn_wmma_f32_16x16x32_bf16(false, aHi[1][kk], false, bHi, (short)0, acc1, false, false);
            acc0 = __builtin_amdgcn_wmma_f32_16x16x32_bf16(false, aHi[0][kk], false, bLo, (short)0, acc0, false, false);
            acc1 = __builtin_amdgcn_wmma_f32_16x16x32_bf16(false, aHi[1][kk], false, bLo, (short)0, acc1, false, false);
            acc0 = __builtin_amdgcn_wmma_f32_16x16x32_bf16(false, aLo[0][kk], false, bHi, (short)0, acc0, false, false);
            acc1 = __builtin_amdgcn_wmma_f32_16x16x32_bf16(false, aLo[1][kk], false, bHi, (short)0, acc1, false, false);
        }

        if (col < CDIM) {  // ragged tail of last chunk: masked lanes contribute 0
#pragma unroll
            for (int r = 0; r < 8; ++r) {
                float x0 = acc0[r] * SCALE_;
                float n0 = fmaxf(m[0][r], x0);
                s[0][r] = s[0][r] * __expf(m[0][r] - n0) + __expf(x0 - n0);
                m[0][r] = n0;
                float x1 = acc1[r] * SCALE_;
                float n1 = fmaxf(m[1][r], x1);
                s[1][r] = s[1][r] * __expf(m[1][r] - n1) + __expf(x1 - n1);
                m[1][r] = n1;
            }
        }
    }

    // ---- reduce the 16 lanes of each half-wave (same row, disjoint columns)
#pragma unroll
    for (int t = 0; t < 2; ++t)
#pragma unroll
        for (int r = 0; r < 8; ++r)
#pragma unroll
            for (int d = 1; d <= 8; d <<= 1) {
                float om = __shfl_xor(m[t][r], d, 32);
                float os = __shfl_xor(s[t][r], d, 32);
                lse_combine(m[t][r], s[t][r], om, os);
            }

    // ---- cross-wave combine via LDS (all 8 waves cover the same 32 rows)
    __shared__ float2 red[32][8];
    if (lm == 0) {
#pragma unroll
        for (int t = 0; t < 2; ++t)
#pragma unroll
            for (int r = 0; r < 8; ++r)
                red[t * 16 + r + 8 * lh][wave] = make_float2(m[t][r], s[t][r]);
    }
    __syncthreads();
    if (threadIdx.x < 32) {
        int rowLoc = threadIdx.x;
        float mm = -1e30f, ss = 0.f;
#pragma unroll
        for (int w = 0; w < 8; ++w) lse_combine(mm, ss, red[rowLoc][w].x, red[rowLoc][w].y);
        partial[(size_t)(rowBase + rowLoc) * NCHUNK + blockIdx.y] = make_float2(mm, ss);
    }
}

// ============================================================
// Kernel 3: per-row finalize — combine chunk partials, exact fp32
//           target dot, ArcFace margin swap, per-row NLL
// ============================================================
__global__ __launch_bounds__(256)
void arc_finish(const float* __restrict__ emb,
                const float* __restrict__ W,
                const int* __restrict__ labels,
                const float* __restrict__ invn,
                const float2* __restrict__ partial,
                float* __restrict__ rowLoss) {
    int row = blockIdx.x * blockDim.x + threadIdx.x;
    if (row >= BDIM) return;
    float mm = -1e30f, ss = 0.f;
    const float2* p = partial + (size_t)row * NCHUNK;
    for (int k = 0; k < NCHUNK; ++k) lse_combine(mm, ss, p[k].x, p[k].y);

    int lab = labels[row];
    float inv = invn[lab];
    const float* e  = emb + (size_t)row * DDIM;
    const float* wc = W + lab;
    float dot = 0.f;
#pragma unroll 8
    for (int d = 0; d < DDIM; ++d) dot = fmaf(e[d], wc[(size_t)d * CDIM], dot);

    float t  = dot * inv;                                  // raw target logit
    float tc = fminf(fmaxf(t, -1.0f + EPS_), 1.0f - EPS_);
    float tm = tc * COS_M - sqrtf(1.0f - tc * tc) * SIN_M; // cos(acos(tc)+M)
    float xr = SCALE_ * t;
    float xm = SCALE_ * tm;
    float nm = fmaxf(mm, xm);
    // swap raw target term for the margin term inside the sum-exp
    float S2 = ss * __expf(mm - nm) - __expf(xr - nm) + __expf(xm - nm);
    rowLoss[row] = __logf(S2) + nm - xm;
}

// ============================================================
// Kernel 4: deterministic mean of 2048 row losses -> d_out[0]
// ============================================================
__global__ __launch_bounds__(256)
void arc_reduce(const float* __restrict__ rowLoss, float* __restrict__ out) {
    __shared__ float sm[8];
    float s = 0.f;
#pragma unroll
    for (int i = 0; i < 8; ++i) s += rowLoss[threadIdx.x + 256 * i];
#pragma unroll
    for (int d = 16; d >= 1; d >>= 1) s += __shfl_xor(s, d, 32);
    if ((threadIdx.x & 31) == 0) sm[threadIdx.x >> 5] = s;
    __syncthreads();
    if (threadIdx.x == 0) {
        float tot = 0.f;
#pragma unroll
        for (int w = 0; w < 8; ++w) tot += sm[w];
        out[0] = tot / (float)BDIM;
    }
}

// ============================================================
extern "C" void kernel_launch(void* const* d_in, const int* in_sizes, int n_in,
                              void* d_out, int out_size, void* d_ws, size_t ws_size,
                              hipStream_t stream) {
    (void)in_sizes; (void)n_in; (void)out_size; (void)ws_size;
    const float* emb    = (const float*)d_in[0];
    const int*   labels = (const int*)  d_in[1];
    const float* W      = (const float*)d_in[2];
    float* out = (float*)d_out;
    char*  ws  = (char*)d_ws;

    unsigned short* wtHi = (unsigned short*)(ws + OFF_HI);
    unsigned short* wtLo = (unsigned short*)(ws + OFF_LO);
    float*  invn    = (float*) (ws + OFF_INVN);
    float2* partial = (float2*)(ws + OFF_PART);
    float*  rowLoss = (float*) (ws + OFF_RLOSS);

    arc_prep<<<dim3((CDIM + 255) / 256), dim3(256), 0, stream>>>(W, wtHi, wtLo, invn);
    arc_gemm_lse<<<dim3(BDIM / 32, NCHUNK), dim3(256), 0, stream>>>(emb, wtHi, wtLo, partial);
    arc_finish<<<dim3(BDIM / 256), dim3(256), 0, stream>>>(emb, W, labels, invn, partial, rowLoss);
    arc_reduce<<<dim3(1), dim3(256), 0, stream>>>(rowLoss, out);
}
